// InternLM2Attention_41162966565473
// MI455X (gfx1250) — matrix operand order repaired and checked by reading, not compiled, not run on hardware
//
#include <hip/hip_runtime.h>
#include <hip/hip_bf16.h>

// ---------------------------------------------------------------------------
// Types for CDNA5 WMMA (wave32): V_WMMA_F32_16X16X32_F16
// ---------------------------------------------------------------------------
typedef __attribute__((ext_vector_type(16))) _Float16 v16h;
typedef __attribute__((ext_vector_type(8)))  _Float16 v8h;
typedef __attribute__((ext_vector_type(8)))  float    v8f;
typedef __attribute__((ext_vector_type(4)))  int      v4i;

#if __has_builtin(__builtin_amdgcn_global_load_async_to_lds_b128) && \
    __has_builtin(__builtin_amdgcn_s_wait_asynccnt)
#define USE_ASYNC_LDS 1
typedef __attribute__((address_space(1))) v4i g_v4i;  // global int32x4
typedef __attribute__((address_space(3))) v4i l_v4i;  // LDS int32x4
#else
#define USE_ASYNC_LDS 0
#endif

union F16Frag { v16h v; v8h h[2]; };

// Load a 16-half fragment as two contiguous 16B chunks separated by gapHalves.
__device__ __forceinline__ v16h ld_frag(const _Float16* p, int gapHalves) {
    F16Frag u;
    u.h[0] = *(const v8h*)(p);
    u.h[1] = *(const v8h*)(p + gapHalves);
    return u.v;
}

__device__ __forceinline__ v8f wmma16(v16h a, v16h b, v8f c) {
    // (neg_a, A, neg_b, B, c_mod, C, reuse_a, reuse_b)
    return __builtin_amdgcn_wmma_f32_16x16x32_f16(false, a, false, b,
                                                  (short)0, c, false, false);
}

// 16-byte copy global -> LDS: async path on CDNA5 (ASYNCcnt), sync fallback.
__device__ __forceinline__ void cp_g2l_16B(_Float16* ldsDst,
                                           const _Float16* gSrc) {
#if USE_ASYNC_LDS
    __builtin_amdgcn_global_load_async_to_lds_b128((g_v4i*)gSrc,
                                                   (l_v4i*)ldsDst, 0, 0);
#else
    *(v8h*)ldsDst = *(const v8h*)gSrc;
#endif
}
__device__ __forceinline__ void waitAsyncLE4() {
#if USE_ASYNC_LDS
    __builtin_amdgcn_s_wait_asynccnt(4);
#endif
}
__device__ __forceinline__ void waitAsyncLE0() {
#if USE_ASYNC_LDS
    __builtin_amdgcn_s_wait_asynccnt(0);
#endif
}

// ---------------------------------------------------------------------------
// f32 -> f16 row-major copy (with destination row stride)
// ---------------------------------------------------------------------------
__global__ void cvt_f32_to_f16_rows(const float* __restrict__ src,
                                    _Float16* __restrict__ dst,
                                    int cols, int ldDst, size_t total) {
    size_t idx = (size_t)blockIdx.x * blockDim.x + threadIdx.x;
    if (idx >= total) return;
    size_t r = idx / (size_t)cols;
    size_t c = idx - r * (size_t)cols;
    dst[r * (size_t)ldDst + c] = (_Float16)src[idx];
}

// ---------------------------------------------------------------------------
// LDS-tiled transpose-convert: src f32 [K][N] -> dst f16 [N][ldDst],
// dst[n*ldDst + kOff + k] = (k<K) ? src[k*N+n] : 0, for k in [0,kPad).
// 64x64 tiles; grid (N/64, ceil(kPad/64)); block 256 (64 x 4).
// Coalesced f32 reads, coalesced 128B f16 writes; 66-half pitch -> no
// LDS bank conflicts on the transposed reads.
// ---------------------------------------------------------------------------
__global__ __launch_bounds__(256) void cvt_transpose_tile(
    const float* __restrict__ src, _Float16* __restrict__ dst,
    int N, int K, int kPad, int ldDst, int kOff) {
    __shared__ _Float16 tile[64][66];
    const int n0 = blockIdx.x * 64, k0 = blockIdx.y * 64;
    const int tx = threadIdx.x & 63, ty = threadIdx.x >> 6;
#pragma unroll 4
    for (int it = 0; it < 16; it++) {
        int k = k0 + it * 4 + ty;
        float v = (k < K) ? src[(size_t)k * N + n0 + tx] : 0.0f;
        tile[it * 4 + ty][tx] = (_Float16)v;  // tile[k_local][n_local]
    }
    __syncthreads();
    const int k = k0 + tx;
    if (k < kPad) {
#pragma unroll 4
        for (int it = 0; it < 16; it++) {
            int nl = it * 4 + ty;
            dst[(size_t)(n0 + nl) * ldDst + kOff + k] = tile[tx][nl];
        }
    }
}

// ---------------------------------------------------------------------------
// Rank-8 QKV LoRA left factor: t1[s][r] = sum_h x[s][h]*la[h][r].
// Writes xA[s][4096 + r] = mask[s] ? 0 : 2.0*t1  (r<8), zeros for r in [8,32).
// ---------------------------------------------------------------------------
__global__ __launch_bounds__(256) void qkv_lora8(const float* __restrict__ x,
                                                 const float* __restrict__ la,
                                                 const int* __restrict__ mask,
                                                 _Float16* __restrict__ xA,
                                                 int lda) {
    __shared__ float red[8][256];
    const int s = blockIdx.x, tid = threadIdx.x;
    float acc[8];
#pragma unroll
    for (int r = 0; r < 8; r++) acc[r] = 0.0f;
    const float* xr = x + (size_t)s * 4096;
    for (int hh = tid; hh < 4096; hh += 256) {
        float xv = xr[hh];
        const float* lar = la + (size_t)hh * 8;
#pragma unroll
        for (int r = 0; r < 8; r++) acc[r] += xv * lar[r];
    }
#pragma unroll
    for (int r = 0; r < 8; r++) red[r][tid] = acc[r];
    __syncthreads();
    for (int off = 128; off > 0; off >>= 1) {
        if (tid < off)
#pragma unroll
            for (int r = 0; r < 8; r++) red[r][tid] += red[r][tid + off];
        __syncthreads();
    }
    if (tid < 32) {
        float v = 0.0f;
        if (tid < 8 && mask[s] == 0) v = 2.0f * red[tid][0];  // WQKV_SCALING
        xA[(size_t)s * lda + 4096 + tid] = (_Float16)v;
    }
}

// ---------------------------------------------------------------------------
// Rank-256 WO LoRA: attnA[s][4096+r] = mask[s] ? 0 : t2[s][r]  (WO_SCALING=1)
// ---------------------------------------------------------------------------
__global__ void t2_mask_store(const float* __restrict__ t2,
                              const int* __restrict__ mask,
                              _Float16* __restrict__ attnA, int lda) {
    int idx = blockIdx.x * blockDim.x + threadIdx.x;
    int s = idx >> 8, r = idx & 255;
    float v = mask[s] ? 0.0f : t2[(size_t)s * 256 + r];
    attnA[(size_t)s * lda + 4096 + r] = (_Float16)v;
}

// ---------------------------------------------------------------------------
// Tiled f16 WMMA GEMM:  C[M x N] = A[M x K] * Bt[N x K]^T
// Block tile 128x128, K-step 32, 8 waves each computing a 32x64 sub-tile.
// Double-buffered LDS; tiles staged with GLOBAL_LOAD_ASYNC_TO_LDS_B128
// (ASYNCcnt-tracked, in-order -> s_wait_asynccnt 4 drains the older tile).
// M = gridDim.y*128, N = gridDim.x*128; K multiple of 32.
// storeHalf: 0 -> f32 C, 1 -> f16 C.
// ---------------------------------------------------------------------------
__global__ __launch_bounds__(256) void gemm_f16_wmma(
    const _Float16* __restrict__ A, int lda,
    const _Float16* __restrict__ Bt, int ldb,
    void* __restrict__ C, int ldc, int K, int storeHalf) {
    __shared__ __align__(16) _Float16 sA[2][128 * 32];
    __shared__ __align__(16) _Float16 sB[2][128 * 32];
    const int bm = blockIdx.y * 128, bn = blockIdx.x * 128;
    const int tid = threadIdx.x;
    const int lane = tid & 31, wave = tid >> 5;
    const int wm = (wave & 3) * 32;   // 4 waves along M
    const int wn = (wave >> 2) * 64;  // 2 waves along N
    const int sub = lane >> 4, l16 = lane & 15;
    // tile-copy coordinates: each thread moves 2x16B of A and 2x16B of B
    const int r0 = tid >> 2, c0 = (tid & 3) * 8;  // rows 0..63
    const int r1 = r0 + 64;                       // rows 64..127

    v8f acc[2][4];
#pragma unroll
    for (int i = 0; i < 2; i++)
#pragma unroll
        for (int j = 0; j < 4; j++)
#pragma unroll
            for (int e = 0; e < 8; e++) acc[i][j][e] = 0.0f;

    auto issueTile = [&](int kt, int b) {
        const size_t ka = (size_t)kt * 32 + c0;
        cp_g2l_16B(&sA[b][r0 * 32 + c0], &A[(size_t)(bm + r0) * lda + ka]);
        cp_g2l_16B(&sA[b][r1 * 32 + c0], &A[(size_t)(bm + r1) * lda + ka]);
        cp_g2l_16B(&sB[b][r0 * 32 + c0], &Bt[(size_t)(bn + r0) * ldb + ka]);
        cp_g2l_16B(&sB[b][r1 * 32 + c0], &Bt[(size_t)(bn + r1) * ldb + ka]);
    };
    auto computeTile = [&](int b) {
        v16h af[2], bf[4];
#pragma unroll
        for (int i = 0; i < 2; i++)  // A 16x32: halves 0..7 -> K=sub*8.., 8..15 -> 16+sub*8..
            af[i] = ld_frag(&sA[b][(wm + i * 16 + l16) * 32 + sub * 8], 16);
#pragma unroll
        for (int j = 0; j < 4; j++)  // B 32x16: halves 0..7 -> K=sub*8.., 8..15 -> 16+sub*8..
            bf[j] = ld_frag(&sB[b][(wn + j * 16 + l16) * 32 + sub * 8], 16);
#pragma unroll
        for (int i = 0; i < 2; i++)
#pragma unroll
            for (int j = 0; j < 4; j++)
                acc[i][j] = wmma16(af[i], bf[j], acc[i][j]);
    };

    const int KT = K >> 5;
    issueTile(0, 0);
    int kt = 0;
    for (; kt < KT - 1; kt++) {
        issueTile(kt + 1, (kt + 1) & 1);  // prefetch next tile into other buf
        waitAsyncLE4();                   // drain the older (current) tile
        __syncthreads();
        computeTile(kt & 1);
        __syncthreads();
    }
    waitAsyncLE0();
    __syncthreads();
    computeTile(kt & 1);

    // C/D layout: VGPR r -> M = r + 8*sub, N = l16
#pragma unroll
    for (int i = 0; i < 2; i++)
#pragma unroll
        for (int j = 0; j < 4; j++) {
            int row0 = bm + wm + i * 16 + sub * 8;
            int col  = bn + wn + j * 16 + l16;
            if (storeHalf) {
                _Float16* Ch = (_Float16*)C;
#pragma unroll
                for (int r = 0; r < 8; r++)
                    Ch[(size_t)(row0 + r) * ldc + col] = (_Float16)acc[i][j][r];
            } else {
                float* Cf = (float*)C;
#pragma unroll
                for (int r = 0; r < 8; r++)
                    Cf[(size_t)(row0 + r) * ldc + col] = acc[i][j][r];
            }
        }
}

// ---------------------------------------------------------------------------
// RoPE + QKV split.  qkvh: [s][6144] f16.
// qh: [h][s][128], kh: [kv][s][128] (rope applied), vT: [kv][d][s] (transposed)
// grid (2048, 48), block 64 (d = 0..63 handles the rotation pair d, d+64)
// ---------------------------------------------------------------------------
__global__ void rope_split(const _Float16* __restrict__ qkv,
                           const int* __restrict__ pos,
                           _Float16* __restrict__ qh,
                           _Float16* __restrict__ kh,
                           _Float16* __restrict__ vT) {
    const int s = blockIdx.x, u = blockIdx.y, d = threadIdx.x;
    const float p = (float)pos[s];
    if (u < 40) {
        const _Float16* src;
        _Float16* dst;
        if (u < 32) {
            src = qkv + (size_t)s * 6144 + u * 128;
            dst = qh + ((size_t)u * 2048 + s) * 128;
        } else {
            int kvh = u - 32;
            src = qkv + (size_t)s * 6144 + 4096 + kvh * 128;
            dst = kh + ((size_t)kvh * 2048 + s) * 128;
        }
        float invf = powf(10000.0f, -(float)d * (1.0f / 64.0f));
        float ang = p * invf;
        float c = cosf(ang), sn = sinf(ang);
        float x1 = (float)src[d], x2 = (float)src[d + 64];
        dst[d]      = (_Float16)(x1 * c - x2 * sn);
        dst[d + 64] = (_Float16)(x2 * c + x1 * sn);
    } else {
        int kvh = u - 40;
        const _Float16* src = qkv + (size_t)s * 6144 + 5120 + kvh * 128;
        vT[((size_t)kvh * 128 + d) * 2048 + s]      = src[d];
        vT[((size_t)kvh * 128 + d + 64) * 2048 + s] = src[d + 64];
    }
}

// ---------------------------------------------------------------------------
// Causal GQA flash attention, HEAD_DIM=128, S=2048, 32 q-heads, 8 kv-heads.
// grid (S/128, 32), block 256.  Each wave privately owns 16 query rows and
// streams keys in chunks of 32 with online softmax; all matmuls are WMMA.
// Output -> attnA[s][h*128 + d] (f16, row stride lda=4352).
// ---------------------------------------------------------------------------
__global__ __launch_bounds__(256) void attn_fa(const _Float16* __restrict__ qh,
                                               const _Float16* __restrict__ kh,
                                               const _Float16* __restrict__ vT,
                                               _Float16* __restrict__ attnA,
                                               int lda) {
    __shared__ __align__(16) _Float16 pbuf[8 * 16 * 32];  // per-wave P tiles
    const int h = blockIdx.y;
    const int kv = h >> 2;  // 4 q-heads per kv-head
    const int wave = threadIdx.x >> 5, lane = threadIdx.x & 31;
    const int sub = lane >> 4, l16 = lane & 15;
    const int q0 = blockIdx.x * 128 + wave * 16;
    _Float16* pw = pbuf + wave * 512;

    // Q fragments: A-layout 16x32 over d-chunks dj=0..3 (loaded once)
    const _Float16* qrow = qh + ((size_t)h * 2048 + q0 + l16) * 128;
    v16h aq[4];
#pragma unroll
    for (int dj = 0; dj < 4; dj++)
        aq[dj] = ld_frag(qrow + dj * 32 + sub * 8, 16);

    float m[8], l[8];
    v8f o[8];
#pragma unroll
    for (int r = 0; r < 8; r++) { m[r] = -1e30f; l[r] = 0.0f; }
#pragma unroll
    for (int dj = 0; dj < 8; dj++)
#pragma unroll
        for (int r = 0; r < 8; r++) o[dj][r] = 0.0f;

    const float scale = 0.088388347648318447f;  // 1/sqrt(128)
    const _Float16* kbp = kh + (size_t)kv * 2048 * 128;  // [key][d]
    const _Float16* vbp = vT + (size_t)kv * 128 * 2048;  // [d][key]
    const int kcMax = (q0 + 15) >> 5;

    for (int kc = 0; kc <= kcMax; kc++) {
        const int kb = kc * 32;
        if (kc < kcMax) {  // hint next K/V tiles into cache
            __builtin_prefetch(kbp + (size_t)(kb + 32 + l16) * 128, 0, 3);
            __builtin_prefetch(vbp + (size_t)l16 * 2048 + kb + 32, 0, 3);
        }
        // scores: two 16x16 tiles (keys kb..kb+15, kb+16..kb+31)
        v8f s0, s1;
#pragma unroll
        for (int r = 0; r < 8; r++) { s0[r] = 0.0f; s1[r] = 0.0f; }
#pragma unroll
        for (int dj = 0; dj < 4; dj++) {
            const _Float16* kp0 =
                kbp + (size_t)(kb + l16) * 128 + dj * 32 + sub * 8;
            v16h b0 = ld_frag(kp0, 16);
            v16h b1 = ld_frag(kp0 + 16 * 128, 16);
            s0 = wmma16(aq[dj], b0, s0);
            s1 = wmma16(aq[dj], b1, s1);
        }
        // scale + causal mask (reference: scale, then -1e9 where not tril)
        float p0[8], p1[8], nm[8], rs[8];
#pragma unroll
        for (int r = 0; r < 8; r++) {
            int row = q0 + sub * 8 + r;
            float x0 = s0[r] * scale; if (kb + l16 > row)      x0 = -1e9f;
            float x1 = s1[r] * scale; if (kb + 16 + l16 > row) x1 = -1e9f;
            p0[r] = x0; p1[r] = x1;
            nm[r] = fmaxf(x0, x1);
        }
        // row-max: each row lives in one 16-lane half -> xor masks 1,2,4,8
#pragma unroll
        for (int off = 1; off < 16; off <<= 1)
#pragma unroll
            for (int r = 0; r < 8; r++)
                nm[r] = fmaxf(nm[r], __shfl_xor(nm[r], off, 32));
#pragma unroll
        for (int r = 0; r < 8; r++) nm[r] = fmaxf(nm[r], m[r]);
#pragma unroll
        for (int r = 0; r < 8; r++) {
            p0[r] = __expf(p0[r] - nm[r]);
            p1[r] = __expf(p1[r] - nm[r]);
            rs[r] = p0[r] + p1[r];
        }
#pragma unroll
        for (int off = 1; off < 16; off <<= 1)
#pragma unroll
            for (int r = 0; r < 8; r++) rs[r] += __shfl_xor(rs[r], off, 32);
#pragma unroll
        for (int r = 0; r < 8; r++) {
            float alpha = __expf(m[r] - nm[r]);
            l[r] = l[r] * alpha + rs[r];
            m[r] = nm[r];
#pragma unroll
            for (int dj = 0; dj < 8; dj++) o[dj][r] *= alpha;
        }
        // C-layout -> A-layout via per-wave private LDS (same-wave, no barrier)
#pragma unroll
        for (int r = 0; r < 8; r++) {
            int rr = sub * 8 + r;
            pw[rr * 32 + l16]      = (_Float16)p0[r];
            pw[rr * 32 + 16 + l16] = (_Float16)p1[r];
        }
        v16h ap = ld_frag(pw + l16 * 32 + sub * 8, 16);
        // O += P(16x32) * V(32x128)
#pragma unroll
        for (int dj = 0; dj < 8; dj++) {
            const _Float16* vp =
                vbp + (size_t)(dj * 16 + l16) * 2048 + kb + sub * 8;
            v16h bv = ld_frag(vp, 16);
            o[dj] = wmma16(ap, bv, o[dj]);
        }
    }
    // normalize and store f16 into attnA[s][h*128 + d]
#pragma unroll
    for (int r = 0; r < 8; r++) {
        float inv = 1.0f / l[r];
        int row = q0 + sub * 8 + r;
#pragma unroll
        for (int dj = 0; dj < 8; dj++)
            attnA[(size_t)row * lda + h * 128 + dj * 16 + l16] =
                (_Float16)(o[dj][r] * inv);
    }
}

// ---------------------------------------------------------------------------
// Host-side orchestration (all on `stream`, no allocations)
// ---------------------------------------------------------------------------
extern "C" void kernel_launch(void* const* d_in, const int* in_sizes, int n_in,
                              void* d_out, int out_size, void* d_ws,
                              size_t ws_size, hipStream_t stream) {
    (void)in_sizes; (void)n_in; (void)out_size; (void)ws_size;
    const float* x         = (const float*)d_in[0];
    const int*   positions = (const int*)d_in[1];
    const int*   im_mask   = (const int*)d_in[2];
    const float* wqkv_w    = (const float*)d_in[3];
    const float* wqkv_la   = (const float*)d_in[4];
    const float* wqkv_lb   = (const float*)d_in[5];
    const float* wo_w      = (const float*)d_in[6];
    const float* wo_la     = (const float*)d_in[7];
    const float* wo_lb     = (const float*)d_in[8];
    float* out = (float*)d_out;

    char* ws = (char*)d_ws;
    size_t off = 0;
    auto alloc = [&](size_t bytes) -> char* {
        char* p = ws + off;
        off = (off + bytes + 255) & ~(size_t)255;
        return p;
    };
    // K-extended A matrices fold the (masked, scaled) LoRA ranks into the GEMM.
    _Float16* xA     = (_Float16*)alloc((size_t)2048 * 4128 * 2);  // [s][4096+32]
    _Float16* wqkvBt = (_Float16*)alloc((size_t)6144 * 4128 * 2);  // [n][4096+32]
    _Float16* qkvh   = (_Float16*)alloc((size_t)2048 * 6144 * 2);
    _Float16* qhp    = (_Float16*)alloc((size_t)32 * 2048 * 128 * 2);
    _Float16* khp    = (_Float16*)alloc((size_t)8 * 2048 * 128 * 2);
    _Float16* vTp    = (_Float16*)alloc((size_t)8 * 128 * 2048 * 2);
    _Float16* attnA  = (_Float16*)alloc((size_t)2048 * 4352 * 2);  // [s][4096+256]
    _Float16* wolaT  = (_Float16*)alloc((size_t)256 * 4096 * 2);
    float*    t2     = (float*)   alloc((size_t)2048 * 256 * 4);
    _Float16* woBt   = (_Float16*)alloc((size_t)4096 * 4352 * 2);  // [n][4096+256]

    // --- Phase 1: QKV projection with folded PLoRA ---
    {
        size_t total = (size_t)2048 * 4096;
        cvt_f32_to_f16_rows<<<(unsigned)((total + 255) / 256), 256, 0, stream>>>(
            x, xA, 4096, 4128, total);
    }
    qkv_lora8<<<2048, 256, 0, stream>>>(x, wqkv_la, im_mask, xA, 4128);
    cvt_transpose_tile<<<dim3(96, 64), 256, 0, stream>>>(
        wqkv_w, wqkvBt, 6144, 4096, 4096, 4128, 0);
    cvt_transpose_tile<<<dim3(96, 1), 256, 0, stream>>>(  // lb: 8 rows -> pad 32
        wqkv_lb, wqkvBt, 6144, 8, 32, 4128, 4096);
    gemm_f16_wmma<<<dim3(6144 / 128, 2048 / 128), 256, 0, stream>>>(
        xA, 4128, wqkvBt, 4128, qkvh, 6144, 4128, 1);

    // --- Phase 2: RoPE + attention ---
    rope_split<<<dim3(2048, 48), 64, 0, stream>>>(qkvh, positions, qhp, khp, vTp);
    attn_fa<<<dim3(2048 / 128, 32), 256, 0, stream>>>(qhp, khp, vTp, attnA, 4352);

    // --- Phase 3: WO projection with folded PLoRA (rank 256) ---
    cvt_transpose_tile<<<dim3(4, 64), 256, 0, stream>>>(
        wo_la, wolaT, 256, 4096, 4096, 4096, 0);
    gemm_f16_wmma<<<dim3(256 / 128, 2048 / 128), 256, 0, stream>>>(
        attnA, 4352, wolaT, 4096, t2, 256, 4096, 0);  // reads only K<4096 cols
    t2_mask_store<<<2048, 256, 0, stream>>>(t2, im_mask, attnA, 4352);
    cvt_transpose_tile<<<dim3(64, 64), 256, 0, stream>>>(
        wo_w, woBt, 4096, 4096, 4096, 4352, 0);
    cvt_transpose_tile<<<dim3(64, 4), 256, 0, stream>>>(
        wo_lb, woBt, 4096, 256, 256, 4352, 4096);
    gemm_f16_wmma<<<dim3(4096 / 128, 2048 / 128), 256, 0, stream>>>(
        attnA, 4352, woBt, 4352, out, 4096, 4352, 0);
}
